// TemporalAttentionLayer_48369921687733
// MI455X (gfx1250) — matrix operand order repaired and checked by reading, not compiled, not run on hardware
//
#include <hip/hip_runtime.h>
#include <hip/hip_bf16.h>

#define TT 128
#define CC 256
#define XS_S 264   // 256 + 8 bf16 pad (16B) to break bank conflicts
#define VT_S 136   // 128 + 8 pad

typedef __attribute__((ext_vector_type(16))) __bf16 bf16x16;
typedef __attribute__((ext_vector_type(8)))  __bf16 bf16x8;
typedef __attribute__((ext_vector_type(8)))  float  f32x8;

// ---- WMMA fragment loaders (bf16 16x16x32, wave32 layouts per ISA 7.12.2) ----
// A 16x32 from row-major [row][col], stride in elements.
// lane l: m = l&15 ; K = kb..kb+7 (elems 0..7) and kb+16..kb+23 (elems 8..15), kb=(l>>4)*8
__device__ __forceinline__ bf16x16 load_frag_a(const __bf16* base, int stride) {
  const int l = threadIdx.x & 31;
  const __bf16* p = base + (l & 15) * stride + ((l >> 4) << 3);
  bf16x8 lo = *(const bf16x8*)p;
  bf16x8 hi = *(const bf16x8*)(p + 16);
  bf16x16 r;
#pragma unroll
  for (int i = 0; i < 8; ++i) { r[i] = lo[i]; r[i + 8] = hi[i]; }
  return r;
}

// B 32x16 from N-major ("transposed") layout Bt[n][k], stride in elements.
// lane l: n = l&15 ; K = kb..kb+7 (elems 0..7) and kb+16..kb+23 (elems 8..15), kb=(l>>4)*8
__device__ __forceinline__ bf16x16 load_frag_b(const __bf16* base, int stride) {
  const int l = threadIdx.x & 31;
  const __bf16* p = base + (l & 15) * stride + ((l >> 4) << 3);
  bf16x8 lo = *(const bf16x8*)p;
  bf16x8 hi = *(const bf16x8*)(p + 16);
  bf16x16 r;
#pragma unroll
  for (int i = 0; i < 8; ++i) { r[i] = lo[i]; r[i + 8] = hi[i]; }
  return r;
}

__device__ __forceinline__ f32x8 wmma_bf16(bf16x16 a, bf16x16 b, f32x8 c) {
  return __builtin_amdgcn_wmma_f32_16x16x32_bf16(false, a, false, b, (short)0, c,
                                                 false, false);
}

// ---- Prepass: fp32 weights -> bf16 in workspace.
// ws layout (elements): [0] WqT 65536 | [65536] WkT | [131072] WvT | [196608] lin_w (as-is)
__global__ void prep_weights_kernel(const float* __restrict__ Wq,
                                    const float* __restrict__ Wk,
                                    const float* __restrict__ Wv,
                                    const float* __restrict__ lw,
                                    __bf16* __restrict__ ws) {
  int idx = blockIdx.x * 256 + threadIdx.x;   // 0 .. 262143
  int mat = idx >> 16;
  int e   = idx & 65535;
  if (mat < 3) {
    const float* src = (mat == 0) ? Wq : (mat == 1) ? Wk : Wv;
    // wT[a][b] = W[b][a]; coalesced writes, strided reads (one-time)
    ws[mat * 65536 + e] = (__bf16)src[(e & 255) * 256 + (e >> 8)];
  } else {
    ws[196608 + e] = (__bf16)lw[e];   // lin_w row-major IS the N-major B for out@lin_w^T
  }
}

// ---- Fused temporal attention layer: one workgroup per sample n ----
__global__ void __launch_bounds__(256, 1)
temporal_attn_kernel(const float* __restrict__ in, const float* __restrict__ lin_b,
                     const __bf16* __restrict__ wqT, const __bf16* __restrict__ wkT,
                     const __bf16* __restrict__ wvT, const __bf16* __restrict__ lwb,
                     float* __restrict__ out) {
  __shared__ __bf16 xs[TT * XS_S];        // x, later reused as attn_out
  __shared__ __bf16 qs[TT * XS_S];
  __shared__ __bf16 ks[TT * XS_S];
  __shared__ __bf16 vsT[CC * VT_S];       // v transposed: [c][t]
  __shared__ __bf16 att[8 * 16 * VT_S];   // per-wave softmax scratch [16][128+pad]

  const int tid = threadIdx.x;
  const int w   = __builtin_amdgcn_readfirstlane(tid >> 5);  // wave id, scalar
  const int l   = tid & 31;
  const int g   = l >> 4;
  const int nn  = l & 15;
  const int n0  = blockIdx.x;
  const float* xin = in + (size_t)n0 * TT * CC;

  // Phase 0: x = input + pos -> bf16 LDS
  for (int i = tid; i < TT * CC / 4; i += 256) {
    int base = i * 4;
    int t = base >> 8, c = base & 255;
    float4 v = *(const float4*)(xin + base);
    float p = (float)t;
    __bf16* d = &xs[t * XS_S + c];
    d[0] = (__bf16)(v.x + p); d[1] = (__bf16)(v.y + p);
    d[2] = (__bf16)(v.z + p); d[3] = (__bf16)(v.w + p);
  }
  __syncthreads();

  // Phase 1: q/k/v = x @ W  (3 x 128 16x16 tiles, K=256; 16 tiles/matrix/wave)
  // q tiles
  for (int j = 0; j < 16; ++j) {
    int tile = w * 16 + j;
    int mt = tile >> 4, nt = tile & 15;
    f32x8 acc = {0, 0, 0, 0, 0, 0, 0, 0};
#pragma unroll
    for (int kt = 0; kt < 8; ++kt) {
      bf16x16 a = load_frag_a(xs + mt * 16 * XS_S + kt * 32, XS_S);
      bf16x16 b = load_frag_b(wqT + nt * 16 * 256 + kt * 32, 256);
      acc = wmma_bf16(a, b, acc);
    }
#pragma unroll
    for (int r = 0; r < 8; ++r)
      qs[(mt * 16 + g * 8 + r) * XS_S + nt * 16 + nn] = (__bf16)acc[r];
  }
  // k tiles
  for (int j = 0; j < 16; ++j) {
    int tile = w * 16 + j;
    int mt = tile >> 4, nt = tile & 15;
    f32x8 acc = {0, 0, 0, 0, 0, 0, 0, 0};
#pragma unroll
    for (int kt = 0; kt < 8; ++kt) {
      bf16x16 a = load_frag_a(xs + mt * 16 * XS_S + kt * 32, XS_S);
      bf16x16 b = load_frag_b(wkT + nt * 16 * 256 + kt * 32, 256);
      acc = wmma_bf16(a, b, acc);
    }
#pragma unroll
    for (int r = 0; r < 8; ++r)
      ks[(mt * 16 + g * 8 + r) * XS_S + nt * 16 + nn] = (__bf16)acc[r];
  }
  // v tiles, stored transposed; per-lane rows are contiguous -> one b128 store
  for (int j = 0; j < 16; ++j) {
    int tile = w * 16 + j;
    int mt = tile >> 4, nt = tile & 15;
    f32x8 acc = {0, 0, 0, 0, 0, 0, 0, 0};
#pragma unroll
    for (int kt = 0; kt < 8; ++kt) {
      bf16x16 a = load_frag_a(xs + mt * 16 * XS_S + kt * 32, XS_S);
      bf16x16 b = load_frag_b(wvT + nt * 16 * 256 + kt * 32, 256);
      acc = wmma_bf16(a, b, acc);
    }
    bf16x8 pk;
#pragma unroll
    for (int r = 0; r < 8; ++r) pk[r] = (__bf16)acc[r];
    *(bf16x8*)&vsT[(nt * 16 + nn) * VT_S + mt * 16 + g * 8] = pk;  // 16B aligned
  }
  __syncthreads();

  // Phase 2: causal attention. 64 tasks = (head h, row-tile mt); 8 per wave.
  const float scale = 0.08838834764831845f;  // 1/sqrt(T), T=128
  for (int jt = 0; jt < 8; ++jt) {
    int task = w * 8 + jt;
    int h = task >> 3, mt = task & 7;

    bf16x16 aq = load_frag_a(qs + mt * 16 * XS_S + h * 32, XS_S);
    f32x8 sc[8];
#pragma unroll
    for (int nt = 0; nt < 8; ++nt) {
      sc[nt] = (f32x8){0, 0, 0, 0, 0, 0, 0, 0};
      if (nt <= mt) {  // wave-uniform: skip fully-masked tiles
        bf16x16 bk = load_frag_b(ks + nt * 16 * XS_S + h * 32, XS_S);
        sc[nt] = wmma_bf16(aq, bk, sc[nt]);
      }
    }
    // scale + causal mask + row max
    float rmax[8];
#pragma unroll
    for (int r = 0; r < 8; ++r) rmax[r] = -1e30f;
#pragma unroll
    for (int nt = 0; nt < 8; ++nt) {
#pragma unroll
      for (int r = 0; r < 8; ++r) {
        int row = mt * 16 + g * 8 + r;
        float s = sc[nt][r] * scale;
        bool valid = (nt < mt) || ((nt == mt) && (nt * 16 + nn <= row));
        s = valid ? s : -1e30f;
        sc[nt][r] = s;
        rmax[r] = fmaxf(rmax[r], s);
      }
    }
#pragma unroll
    for (int r = 0; r < 8; ++r) {  // reduce over the 16 lanes holding each row
      rmax[r] = fmaxf(rmax[r], __shfl_xor(rmax[r], 1, 32));
      rmax[r] = fmaxf(rmax[r], __shfl_xor(rmax[r], 2, 32));
      rmax[r] = fmaxf(rmax[r], __shfl_xor(rmax[r], 4, 32));
      rmax[r] = fmaxf(rmax[r], __shfl_xor(rmax[r], 8, 32));
    }
    float rsum[8];
#pragma unroll
    for (int r = 0; r < 8; ++r) rsum[r] = 0.f;
#pragma unroll
    for (int nt = 0; nt < 8; ++nt) {
#pragma unroll
      for (int r = 0; r < 8; ++r) {
        float e = __expf(sc[nt][r] - rmax[r]);
        sc[nt][r] = e;
        rsum[r] += e;
      }
    }
#pragma unroll
    for (int r = 0; r < 8; ++r) {
      rsum[r] += __shfl_xor(rsum[r], 1, 32);
      rsum[r] += __shfl_xor(rsum[r], 2, 32);
      rsum[r] += __shfl_xor(rsum[r], 4, 32);
      rsum[r] += __shfl_xor(rsum[r], 8, 32);
      rsum[r] = __frcp_rn(rsum[r]);
    }
    // attn -> bf16 scratch (re-fragment C/D layout into A layout via LDS)
    __bf16* aw = att + w * 16 * VT_S;
#pragma unroll
    for (int nt = 0; nt < 8; ++nt)
#pragma unroll
      for (int r = 0; r < 8; ++r)
        aw[(g * 8 + r) * VT_S + nt * 16 + nn] = (__bf16)(sc[nt][r] * rsum[r]);

    // out_h = attn @ v  (K = up to mt*16+16, zero beyond)
#pragma unroll
    for (int dt = 0; dt < 2; ++dt) {
      f32x8 acc = {0, 0, 0, 0, 0, 0, 0, 0};
#pragma unroll
      for (int kt = 0; kt < 4; ++kt) {
        if (kt <= (mt >> 1)) {  // wave-uniform causal skip
          bf16x16 a = load_frag_a(aw + kt * 32, VT_S);
          bf16x16 b = load_frag_b(vsT + (h * 32 + dt * 16) * VT_S + kt * 32, VT_S);
          acc = wmma_bf16(a, b, acc);
        }
      }
#pragma unroll
      for (int r = 0; r < 8; ++r)
        xs[(mt * 16 + g * 8 + r) * XS_S + h * 32 + dt * 16 + nn] = (__bf16)acc[r];
    }
  }
  __syncthreads();

  // Phase 3: ff = relu(out @ lin_w^T + b) + out + x ; 128 tiles, 16 per wave
  for (int j = 0; j < 16; ++j) {
    int tile = w * 16 + j;
    int mt = tile >> 4, nt = tile & 15;
    f32x8 acc = {0, 0, 0, 0, 0, 0, 0, 0};
#pragma unroll
    for (int kt = 0; kt < 8; ++kt) {
      bf16x16 a = load_frag_a(xs + mt * 16 * XS_S + kt * 32, XS_S);
      bf16x16 b = load_frag_b(lwb + nt * 16 * 256 + kt * 32, 256);
      acc = wmma_bf16(a, b, acc);
    }
    int col = nt * 16 + nn;
    float bias = lin_b[col];
#pragma unroll
    for (int r = 0; r < 8; ++r) {
      int row = mt * 16 + g * 8 + r;
      float ao = (float)xs[row * XS_S + col];                 // attn out
      float xv = xin[row * CC + col] + (float)row;            // x = input + pos
      float v = fmaxf(acc[r] + bias, 0.f) + ao + xv;
      out[(size_t)n0 * TT * CC + row * CC + col] = v;
    }
  }
}

extern "C" void kernel_launch(void* const* d_in, const int* in_sizes, int n_in,
                              void* d_out, int out_size, void* d_ws, size_t ws_size,
                              hipStream_t stream) {
  const float* in = (const float*)d_in[0];
  const float* Wq = (const float*)d_in[1];
  const float* Wk = (const float*)d_in[2];
  const float* Wv = (const float*)d_in[3];
  const float* lw = (const float*)d_in[4];
  const float* lb = (const float*)d_in[5];
  __bf16* ws = (__bf16*)d_ws;

  prep_weights_kernel<<<1024, 256, 0, stream>>>(Wq, Wk, Wv, lw, ws);
  temporal_attn_kernel<<<2048, 256, 0, stream>>>(
      in, lb, ws, ws + 65536, ws + 131072, ws + 196608, (float*)d_out);
}
